// SuperpositionDistillerV2_63204738728230
// MI455X (gfx1250) — 16-bit and 8-bit matrix operand order repaired and checked by reading, compiled, not run on hardware
//
#include <hip/hip_runtime.h>
#include <hip/hip_bf16.h>
#include <stdint.h>

#define HIDD   2048
#define NHEADS 16
#define MLPDIM 8192
#define SEQL   2048
#define NBATCH 2
#define MROWS  (NBATCH * SEQL)
#define HD     128

typedef __attribute__((ext_vector_type(8)))  int      v8i;
typedef __attribute__((ext_vector_type(8)))  float    v8f;
typedef __attribute__((ext_vector_type(16))) _Float16 v16h;
typedef __attribute__((ext_vector_type(4)))  unsigned int u32x4;
typedef __attribute__((ext_vector_type(4)))  int      i32x4;
typedef __attribute__((ext_vector_type(8)))  int      i32x8;
typedef _Float16 f16;

#if __has_builtin(__builtin_amdgcn_tensor_load_to_lds)
#define HAVE_TDM 1
#else
#define HAVE_TDM 0
#endif

static __device__ __forceinline__ float warp16_max(float v) {
#pragma unroll
  for (int off = 1; off < 16; off <<= 1) v = fmaxf(v, __shfl_xor(v, off, 32));
  return v;
}
static __device__ __forceinline__ float warp16_sum(float v) {
#pragma unroll
  for (int off = 1; off < 16; off <<= 1) v += __shfl_xor(v, off, 32);
  return v;
}

static __device__ __forceinline__ uint32_t lds_offset_of(const void* p) {
  return (uint32_t)(uintptr_t)(__attribute__((address_space(3))) const void*)p;
}

static __device__ __forceinline__ void wait_tensorcnt0() {
#if __has_builtin(__builtin_amdgcn_s_wait_tensorcnt)
  __builtin_amdgcn_s_wait_tensorcnt(0);
#else
  asm volatile("s_wait_tensorcnt 0" ::: "memory");
#endif
}

static __device__ __forceinline__ void tdm_load_tile_2d(
    uint32_t lds_off, const void* gptr, uint32_t tile_bytes, uint32_t tile_rows,
    uint32_t stride_bytes, uint32_t tensor_x, uint32_t tensor_y) {
#if HAVE_TDM
  const uint64_t ga = (uint64_t)(uintptr_t)gptr;
  u32x4 g0;
  g0[0] = 1u;
  g0[1] = lds_off;
  g0[2] = (uint32_t)ga;
  g0[3] = (uint32_t)((ga >> 32) & 0x01FFFFFFu)
          | (2u << 30);
  i32x8 g1;
  g1[0] = 0;
  g1[1] = (int)((tensor_x & 0xFFFFu) << 16);
  g1[2] = (int)((tensor_x >> 16) | ((tensor_y & 0xFFFFu) << 16));
  g1[3] = (int)((tensor_y >> 16) | (tile_bytes << 16));
  g1[4] = (int)tile_rows;
  g1[5] = (int)stride_bytes;
  g1[6] = 0;
  g1[7] = 0;
  i32x4 z4 = {};
#if __clang_major__ >= 23
  i32x8 z8 = {};
  __builtin_amdgcn_tensor_load_to_lds(g0, g1, z4, z4, z8, 0);
#else
  __builtin_amdgcn_tensor_load_to_lds(g0, g1, z4, z4, 0);
#endif
#else
  (void)lds_off; (void)gptr; (void)tile_bytes; (void)tile_rows;
  (void)stride_bytes; (void)tensor_x; (void)tensor_y;
#endif
}

__global__ void __launch_bounds__(256) quant_weight_kernel(
    const float* __restrict__ W, int8_t* __restrict__ Wq,
    float* __restrict__ wscale, int K) {
  const int row = blockIdx.x, t = threadIdx.x;
  __shared__ float red[256];
  float s = 0.f;
  for (int i = t; i < K; i += 256) s += fabsf(W[(size_t)row * K + i]);
  red[t] = s;
  __syncthreads();
  for (int st = 128; st > 0; st >>= 1) {
    if (t < st) red[t] += red[t + st];
    __syncthreads();
  }
  const float scale = red[0] / (float)K + 1e-8f;
  if (t == 0) wscale[row] = scale;
  const float inv = 1.0f / scale;
  for (int i = t; i < K; i += 256) {
    float q = rintf(W[(size_t)row * K + i] * inv);
    q = fmaxf(-1.f, fminf(1.f, q));
    Wq[(size_t)row * K + i] = (int8_t)(int)q;
  }
}

__global__ void __launch_bounds__(256) rmsnorm_quant_kernel(
    const float* __restrict__ X, const float* __restrict__ nw,
    int8_t* __restrict__ Xq, float* __restrict__ ascale) {
  const int row = blockIdx.x, t = threadIdx.x;
  const float* x = X + (size_t)row * HIDD;
  float loc[8];
  float ss = 0.f;
#pragma unroll
  for (int i = 0; i < 8; ++i) {
    loc[i] = x[t + i * 256];
    ss += loc[i] * loc[i];
  }
  __shared__ float red[256];
  red[t] = ss;
  __syncthreads();
  for (int st = 128; st > 0; st >>= 1) {
    if (t < st) red[t] += red[t + st];
    __syncthreads();
  }
  const float rinv = 1.0f / sqrtf(red[0] / (float)HIDD + 1e-6f);
  __syncthreads();
  float amax = 0.f;
#pragma unroll
  for (int i = 0; i < 8; ++i) {
    loc[i] = loc[i] * rinv * nw[t + i * 256];
    amax = fmaxf(amax, fabsf(loc[i]));
  }
  red[t] = amax;
  __syncthreads();
  for (int st = 128; st > 0; st >>= 1) {
    if (t < st) red[t] = fmaxf(red[t], red[t + st]);
    __syncthreads();
  }
  const float as = red[0] + 1e-8f;
  if (t == 0) ascale[row] = as;
  const float q127 = 127.0f / as;
#pragma unroll
  for (int i = 0; i < 8; ++i) {
    float q = rintf(loc[i] * q127);
    q = fmaxf(-128.f, fminf(127.f, q));
    Xq[(size_t)row * HIDD + t + i * 256] = (int8_t)(int)q;
  }
}

template <int PER>
__global__ void __launch_bounds__(256) quant_rows_kernel(
    const float* __restrict__ X, int8_t* __restrict__ Xq,
    float* __restrict__ ascale) {
  const int row = blockIdx.x, t = threadIdx.x;
  const int D = PER * 256;
  const float* x = X + (size_t)row * D;
  float loc[PER];
  float amax = 0.f;
#pragma unroll
  for (int i = 0; i < PER; ++i) {
    loc[i] = x[t + i * 256];
    amax = fmaxf(amax, fabsf(loc[i]));
  }
  __shared__ float red[256];
  red[t] = amax;
  __syncthreads();
  for (int st = 128; st > 0; st >>= 1) {
    if (t < st) red[t] = fmaxf(red[t], red[t + st]);
    __syncthreads();
  }
  const float as = red[0] + 1e-8f;
  if (t == 0) ascale[row] = as;
  const float q127 = 127.0f / as;
#pragma unroll
  for (int i = 0; i < PER; ++i) {
    float q = rintf(loc[i] * q127);
    q = fmaxf(-128.f, fminf(127.f, q));
    Xq[(size_t)row * D + t + i * 256] = (int8_t)(int)q;
  }
}

template <int EPI>
__global__ void __launch_bounds__(256) gemm_iu8_kernel(
    const int8_t* __restrict__ Aq, const float* __restrict__ ascale,
    const int8_t* __restrict__ Wq, const float* __restrict__ wscale,
    const float* __restrict__ alpha, const float* __restrict__ resid,
    void* __restrict__ Yout, int M, int N, int K) {
  __shared__ __align__(16) int8_t As[64][64];
  __shared__ __align__(16) int8_t Bs[64][64];
  const int t = threadIdx.x;
  const int wave = t >> 5, lane = t & 31;
  const int mw = wave & 3, nw = wave >> 2;
  const int n0 = blockIdx.x * 64, m0 = blockIdx.y * 64;
  const int hlf = lane >> 4, l16 = lane & 15;
  const int lrow = t >> 2, lcol = (t & 3) * 16;

  v8i c0 = {}, c1 = {};
  for (int k0 = 0; k0 < K; k0 += 64) {
    __syncthreads();
#if HAVE_TDM
    if (wave == 0) {
      tdm_load_tile_2d(lds_offset_of(&As[0][0]),
                       &Aq[(size_t)m0 * K + k0], 64, 64, (uint32_t)K,
                       (uint32_t)K, 1u << 20);
      tdm_load_tile_2d(lds_offset_of(&Bs[0][0]),
                       &Wq[(size_t)n0 * K + k0], 64, 64, (uint32_t)K,
                       (uint32_t)K, 1u << 20);
      wait_tensorcnt0();
    }
#else
    *(int4*)&As[lrow][lcol] =
        *(const int4*)&Aq[(size_t)(m0 + lrow) * K + k0 + lcol];
    *(int4*)&Bs[lrow][lcol] =
        *(const int4*)&Wq[(size_t)(n0 + lrow) * K + k0 + lcol];
    if (k0 + 64 < K) {
      __builtin_prefetch(&Aq[(size_t)(m0 + lrow) * K + k0 + 64 + lcol], 0, 1);
      __builtin_prefetch(&Wq[(size_t)(n0 + lrow) * K + k0 + 64 + lcol], 0, 1);
    }
#endif
    __syncthreads();

    union {
      v8i v;
      int i[8];
    } a, b0, b1;
#pragma unroll
    for (int j = 0; j < 8; ++j) {
      const int kl = ((j >> 1) << 4) + (hlf << 3) + ((j & 1) << 2);
      a.i[j] = *(const int*)&As[mw * 16 + l16][kl];
    }
#pragma unroll
    for (int j = 0; j < 8; ++j) {
      const int kl = ((j >> 1) << 4) + (hlf << 3) + ((j & 1) << 2);
      b0.i[j] = *(const int*)&Bs[nw * 32 + l16][kl];
      b1.i[j] = *(const int*)&Bs[nw * 32 + 16 + l16][kl];
    }
    c0 = __builtin_amdgcn_wmma_i32_16x16x64_iu8(true, a.v, true, b0.v, c0,
                                                false, false);
    c1 = __builtin_amdgcn_wmma_i32_16x16x64_iu8(true, a.v, true, b1.v, c1,
                                                false, false);
  }

#pragma unroll
  for (int r = 0; r < 8; ++r) {
    const int mg = m0 + mw * 16 + hlf * 8 + r;
    const float asc = ascale[mg] * (1.0f / 127.0f);
#pragma unroll
    for (int f = 0; f < 2; ++f) {
      const int ng = n0 + nw * 32 + f * 16 + l16;
      const int ci = (f == 0) ? c0[r] : c1[r];
      float y = (float)ci * wscale[ng] * asc * alpha[ng];
      const size_t idx = (size_t)mg * N + ng;
      if constexpr (EPI == 0) {
        ((float*)Yout)[idx] = y;
      } else if constexpr (EPI == 1) {
        ((f16*)Yout)[idx] = (f16)y;
      } else {
        ((float*)Yout)[idx] = resid[idx] + y;
      }
    }
  }
}

__global__ void __launch_bounds__(256) gemm_iu8_gated_kernel(
    const int8_t* __restrict__ Aq, const float* __restrict__ ascale,
    const int8_t* __restrict__ Gq, const float* __restrict__ gscale,
    const float* __restrict__ galpha, const int8_t* __restrict__ Uq,
    const float* __restrict__ uscale, const float* __restrict__ ualpha,
    float* __restrict__ Act, int M, int N, int K) {
  __shared__ __align__(16) int8_t As[64][64];
  __shared__ __align__(16) int8_t Gs[64][64];
  __shared__ __align__(16) int8_t Us[64][64];
  const int t = threadIdx.x;
  const int wave = t >> 5, lane = t & 31;
  const int mw = wave & 3, nw = wave >> 2;
  const int n0 = blockIdx.x * 64, m0 = blockIdx.y * 64;
  const int hlf = lane >> 4, l16 = lane & 15;
  const int lrow = t >> 2, lcol = (t & 3) * 16;

  v8i cg0 = {}, cg1 = {}, cu0 = {}, cu1 = {};
  for (int k0 = 0; k0 < K; k0 += 64) {
    __syncthreads();
#if HAVE_TDM
    if (wave == 0) {
      tdm_load_tile_2d(lds_offset_of(&As[0][0]),
                       &Aq[(size_t)m0 * K + k0], 64, 64, (uint32_t)K,
                       (uint32_t)K, 1u << 20);
      tdm_load_tile_2d(lds_offset_of(&Gs[0][0]),
                       &Gq[(size_t)n0 * K + k0], 64, 64, (uint32_t)K,
                       (uint32_t)K, 1u << 20);
      tdm_load_tile_2d(lds_offset_of(&Us[0][0]),
                       &Uq[(size_t)n0 * K + k0], 64, 64, (uint32_t)K,
                       (uint32_t)K, 1u << 20);
      wait_tensorcnt0();
    }
#else
    *(int4*)&As[lrow][lcol] =
        *(const int4*)&Aq[(size_t)(m0 + lrow) * K + k0 + lcol];
    *(int4*)&Gs[lrow][lcol] =
        *(const int4*)&Gq[(size_t)(n0 + lrow) * K + k0 + lcol];
    *(int4*)&Us[lrow][lcol] =
        *(const int4*)&Uq[(size_t)(n0 + lrow) * K + k0 + lcol];
#endif
    __syncthreads();

    union {
      v8i v;
      int i[8];
    } a, g0, g1, u0, u1;
#pragma unroll
    for (int j = 0; j < 8; ++j) {
      const int kl = ((j >> 1) << 4) + (hlf << 3) + ((j & 1) << 2);
      a.i[j] = *(const int*)&As[mw * 16 + l16][kl];
    }
#pragma unroll
    for (int j = 0; j < 8; ++j) {
      const int kl = ((j >> 1) << 4) + (hlf << 3) + ((j & 1) << 2);
      g0.i[j] = *(const int*)&Gs[nw * 32 + l16][kl];
      g1.i[j] = *(const int*)&Gs[nw * 32 + 16 + l16][kl];
      u0.i[j] = *(const int*)&Us[nw * 32 + l16][kl];
      u1.i[j] = *(const int*)&Us[nw * 32 + 16 + l16][kl];
    }
    cg0 = __builtin_amdgcn_wmma_i32_16x16x64_iu8(true, a.v, true, g0.v, cg0,
                                                 false, false);
    cg1 = __builtin_amdgcn_wmma_i32_16x16x64_iu8(true, a.v, true, g1.v, cg1,
                                                 false, false);
    cu0 = __builtin_amdgcn_wmma_i32_16x16x64_iu8(true, a.v, true, u0.v, cu0,
                                                 false, false);
    cu1 = __builtin_amdgcn_wmma_i32_16x16x64_iu8(true, a.v, true, u1.v, cu1,
                                                 false, false);
  }

#pragma unroll
  for (int r = 0; r < 8; ++r) {
    const int mg = m0 + mw * 16 + hlf * 8 + r;
    const float asc = ascale[mg] * (1.0f / 127.0f);
#pragma unroll
    for (int f = 0; f < 2; ++f) {
      const int ng = n0 + nw * 32 + f * 16 + l16;
      const int gi = (f == 0) ? cg0[r] : cg1[r];
      const int ui = (f == 0) ? cu0[r] : cu1[r];
      const float g = (float)gi * gscale[ng] * asc * galpha[ng];
      const float u = (float)ui * uscale[ng] * asc * ualpha[ng];
      const float silu = g / (1.0f + __expf(-g));
      Act[(size_t)mg * N + ng] = silu * u;
    }
  }
}

__global__ void __launch_bounds__(128) rope_pack_kernel(
    const f16* __restrict__ qr, const f16* __restrict__ kr,
    const f16* __restrict__ vr, f16* __restrict__ qo, f16* __restrict__ ko,
    f16* __restrict__ vo) {
  const int idx = blockIdx.x;
  const int s = idx % SEQL;
  const int bh = idx / SEQL;
  const int h = bh % NHEADS;
  const int b = bh / NHEADS;
  const int d = threadIdx.x;
  const size_t src = ((size_t)(b * SEQL + s) * HIDD) + h * HD + d;
  const size_t dst = ((size_t)bh * SEQL + s) * HD + d;
  const int dr = d & 63;
  const float inv_freq = __powf(10000.0f, -(float)(2 * dr) / (float)HD);
  const float ang = (float)s * inv_freq;
  const float c = __cosf(ang), si = __sinf(ang);
  const size_t srcp = (d < 64) ? src + 64 : src - 64;
  const float sgn = (d < 64) ? -1.f : 1.f;
  qo[dst] = (f16)((float)qr[src] * c + sgn * (float)qr[srcp] * si);
  ko[dst] = (f16)((float)kr[src] * c + sgn * (float)kr[srcp] * si);
  vo[dst] = vr[src];
}

__global__ void __launch_bounds__(256) attn_kernel(
    const f16* __restrict__ Q, const f16* __restrict__ Kp,
    const f16* __restrict__ V, const int* __restrict__ amask,
    float* __restrict__ Out) {
  __shared__ __align__(16) f16 Ks[32][128];
  __shared__ __align__(16) f16 Vt[128][32];
  __shared__ __align__(16) f16 Pw[8][16][32];

  const int t = threadIdx.x;
  const int wave = t >> 5, lane = t & 31;
  const int hlf = lane >> 4, l16 = lane & 15;
  const int bh = blockIdx.y;
  const int b = bh / NHEADS, hh = bh % NHEADS;
  const int qt0 = blockIdx.x * 128;
  const size_t base = (size_t)bh * SEQL * HD;
  const int* am = amask + b * SEQL;
  const float sm_scale = 0.08838834764831845f;

  union AF {
    v16h h;
    uint32_t u[8];
  };
  const int qrow0 = qt0 + wave * 16;
  AF qf[4];
#pragma unroll
  for (int c = 0; c < 4; ++c)
#pragma unroll
    for (int e = 0; e < 8; ++e) {
      const int k = 32 * c + ((e >> 2) << 4) + (hlf << 3) + ((e & 3) << 1);
      qf[c].u[e] = *(const uint32_t*)&Q[base + (size_t)(qrow0 + l16) * HD + k];
    }

  v8f o[8];
#pragma unroll
  for (int i = 0; i < 8; ++i) o[i] = {};
  float mrow[8], lrow[8];
#pragma unroll
  for (int r = 0; r < 8; ++r) {
    mrow[r] = -1e30f;
    lrow[r] = 0.f;
  }

  const int kend = qt0 + 128;
  for (int kt = 0; kt < kend; kt += 32) {
    __syncthreads();
    {
      const int key = t >> 3, seg = t & 7;
      const f16* ksrc = &Kp[base + (size_t)(kt + key) * HD + seg * 16];
      const uint32_t klds = lds_offset_of(&Ks[key][seg * 16]);
      asm volatile("global_load_async_to_lds_b128 %0, %1, off"
                   :: "v"(klds), "v"(ksrc) : "memory");
      asm volatile("global_load_async_to_lds_b128 %0, %1, off offset:16"
                   :: "v"(klds), "v"(ksrc) : "memory");
      const f16* vsrc = &V[base + (size_t)(kt + key) * HD + seg * 16];
      f16 vtmp[16];
      *(uint4*)&vtmp[0] = *(const uint4*)vsrc;
      *(uint4*)&vtmp[8] = *(const uint4*)(vsrc + 8);
#pragma unroll
      for (int i = 0; i < 16; ++i) Vt[seg * 16 + i][key] = vtmp[i];
      asm volatile("s_wait_asynccnt 0" ::: "memory");
    }
    __syncthreads();

    v8f s0 = {}, s1 = {};
#pragma unroll
    for (int c = 0; c < 4; ++c) {
      AF b0, b1;
#pragma unroll
      for (int e = 0; e < 8; ++e) {
        const int d = 32 * c + ((e >> 2) << 4) + (hlf << 3) + ((e & 3) << 1);
        b0.u[e] = *(const uint32_t*)&Ks[l16][d];
        b1.u[e] = *(const uint32_t*)&Ks[16 + l16][d];
      }
      s0 = __builtin_amdgcn_wmma_f32_16x16x32_f16(false, qf[c].h, false, b0.h,
                                                  (short)0, s0, false, false);
      s1 = __builtin_amdgcn_wmma_f32_16x16x32_f16(false, qf[c].h, false, b1.h,
                                                  (short)0, s1, false, false);
    }

    const int key0 = kt + l16, key1 = key0 + 16;
    const int ms0 = am[key0], ms1 = am[key1];
    float p0a[8], p1a[8], corr[8];
#pragma unroll
    for (int r = 0; r < 8; ++r) {
      const int qg = qrow0 + hlf * 8 + r;
      float v0 = s0[r] * sm_scale;
      float v1 = s1[r] * sm_scale;
      if (key0 > qg || ms0 == 0) v0 = -1e4f;
      if (key1 > qg || ms1 == 0) v1 = -1e4f;
      const float rm = warp16_max(fmaxf(v0, v1));
      const float nm = fmaxf(mrow[r], rm);
      const float cr = __expf(mrow[r] - nm);
      const float e0 = __expf(v0 - nm), e1 = __expf(v1 - nm);
      const float rs = warp16_sum(e0 + e1);
      lrow[r] = lrow[r] * cr + rs;
      mrow[r] = nm;
      corr[r] = cr;
      p0a[r] = e0;
      p1a[r] = e1;
    }
#pragma unroll
    for (int nc = 0; nc < 8; ++nc)
#pragma unroll
      for (int r = 0; r < 8; ++r) o[nc][r] *= corr[r];

#pragma unroll
    for (int r = 0; r < 8; ++r) {
      Pw[wave][hlf * 8 + r][l16] = (f16)p0a[r];
      Pw[wave][hlf * 8 + r][16 + l16] = (f16)p1a[r];
    }
    __builtin_amdgcn_wave_barrier();
    asm volatile("s_wait_dscnt 0" ::: "memory");
    AF pf;
#pragma unroll
    for (int e = 0; e < 8; ++e) {
      const int k = ((e >> 2) << 4) + (hlf << 3) + ((e & 3) << 1);
      pf.u[e] = *(const uint32_t*)&Pw[wave][l16][k];
    }

#pragma unroll
    for (int nc = 0; nc < 8; ++nc) {
      AF bf;
#pragma unroll
      for (int e = 0; e < 8; ++e)
        bf.u[e] = *(const uint32_t*)&Vt[nc * 16 + l16][((e >> 2) << 4) + (hlf << 3) + ((e & 3) << 1)];
      o[nc] = __builtin_amdgcn_wmma_f32_16x16x32_f16(false, pf.h, false, bf.h,
                                                     (short)0, o[nc], false,
                                                     false);
    }
  }

#pragma unroll
  for (int r = 0; r < 8; ++r) {
    const int qg = qrow0 + hlf * 8 + r;
    const float inv = (lrow[r] > 0.f) ? (1.0f / lrow[r]) : 0.f;
#pragma unroll
    for (int nc = 0; nc < 8; ++nc)
      Out[(size_t)(b * SEQL + qg) * HIDD + hh * HD + nc * 16 + l16] =
          o[nc][r] * inv;
  }
}

extern "C" void kernel_launch(void* const* d_in, const int* in_sizes, int n_in,
                              void* d_out, int out_size, void* d_ws,
                              size_t ws_size, hipStream_t stream) {
  (void)in_sizes;
  (void)n_in;
  (void)out_size;
  (void)ws_size;
  const float* x = (const float*)d_in[0];
  const int* amask = (const int*)d_in[1];
  const float* qw = (const float*)d_in[2];
  const float* qa = (const float*)d_in[3];
  const float* kw = (const float*)d_in[4];
  const float* ka = (const float*)d_in[5];
  const float* vw = (const float*)d_in[6];
  const float* va = (const float*)d_in[7];
  const float* ow = (const float*)d_in[8];
  const float* oa = (const float*)d_in[9];
  const float* gw = (const float*)d_in[10];
  const float* ga = (const float*)d_in[11];
  const float* uw = (const float*)d_in[12];
  const float* ua = (const float*)d_in[13];
  const float* dw = (const float*)d_in[14];
  const float* da = (const float*)d_in[15];
  const float* n1 = (const float*)d_in[16];
  const float* n2 = (const float*)d_in[17];

  char* ws = (char*)d_ws;
  size_t off = 0;
  auto alloc = [&](size_t bytes) -> void* {
    void* p = ws + off;
    off += (bytes + 255) & ~(size_t)255;
    return p;
  };

  int8_t* wqq = (int8_t*)alloc((size_t)HIDD * HIDD);
  int8_t* wqk = (int8_t*)alloc((size_t)HIDD * HIDD);
  int8_t* wqv = (int8_t*)alloc((size_t)HIDD * HIDD);
  int8_t* wqo = (int8_t*)alloc((size_t)HIDD * HIDD);
  float* sq = (float*)alloc(HIDD * 4);
  float* sk = (float*)alloc(HIDD * 4);
  float* sv = (float*)alloc(HIDD * 4);
  float* so = (float*)alloc(HIDD * 4);
  int8_t* wqg = (int8_t*)alloc((size_t)MLPDIM * HIDD);
  int8_t* wqu = (int8_t*)alloc((size_t)MLPDIM * HIDD);
  float* sg = (float*)alloc(MLPDIM * 4);
  float* su = (float*)alloc(MLPDIM * 4);
  int8_t* wqd = (int8_t*)alloc((size_t)HIDD * MLPDIM);
  float* sd = (float*)alloc(HIDD * 4);

  int8_t* hq = (int8_t*)alloc((size_t)MROWS * HIDD);
  float* as1 = (float*)alloc(MROWS * 4);
  f16* qraw = (f16*)alloc((size_t)MROWS * HIDD * 2);
  f16* kraw = (f16*)alloc((size_t)MROWS * HIDD * 2);
  f16* vraw = (f16*)alloc((size_t)MROWS * HIDD * 2);
  f16* qh = (f16*)alloc((size_t)MROWS * HIDD * 2);
  f16* kh = (f16*)alloc((size_t)MROWS * HIDD * 2);
  f16* vh = (f16*)alloc((size_t)MROWS * HIDD * 2);
  float* attn_out = (float*)alloc((size_t)MROWS * HIDD * 4);
  int8_t* aoq = (int8_t*)alloc((size_t)MROWS * HIDD);
  float* aso = (float*)alloc(MROWS * 4);
  float* x1 = (float*)alloc((size_t)MROWS * HIDD * 4);
  int8_t* h2q = (int8_t*)alloc((size_t)MROWS * HIDD);
  float* as2 = (float*)alloc(MROWS * 4);
  float* act = (float*)alloc((size_t)MROWS * MLPDIM * 4);
  int8_t* actq = (int8_t*)alloc((size_t)MROWS * MLPDIM);
  float* as3 = (float*)alloc(MROWS * 4);

  quant_weight_kernel<<<HIDD, 256, 0, stream>>>(qw, wqq, sq, HIDD);
  quant_weight_kernel<<<HIDD, 256, 0, stream>>>(kw, wqk, sk, HIDD);
  quant_weight_kernel<<<HIDD, 256, 0, stream>>>(vw, wqv, sv, HIDD);
  quant_weight_kernel<<<HIDD, 256, 0, stream>>>(ow, wqo, so, HIDD);
  quant_weight_kernel<<<MLPDIM, 256, 0, stream>>>(gw, wqg, sg, HIDD);
  quant_weight_kernel<<<MLPDIM, 256, 0, stream>>>(uw, wqu, su, HIDD);
  quant_weight_kernel<<<HIDD, 256, 0, stream>>>(dw, wqd, sd, MLPDIM);

  rmsnorm_quant_kernel<<<MROWS, 256, 0, stream>>>(x, n1, hq, as1);

  dim3 g1(HIDD / 64, MROWS / 64);
  gemm_iu8_kernel<1><<<g1, 256, 0, stream>>>(hq, as1, wqq, sq, qa, nullptr,
                                             qraw, MROWS, HIDD, HIDD);
  gemm_iu8_kernel<1><<<g1, 256, 0, stream>>>(hq, as1, wqk, sk, ka, nullptr,
                                             kraw, MROWS, HIDD, HIDD);
  gemm_iu8_kernel<1><<<g1, 256, 0, stream>>>(hq, as1, wqv, sv, va, nullptr,
                                             vraw, MROWS, HIDD, HIDD);

  rope_pack_kernel<<<NBATCH * NHEADS * SEQL, 128, 0, stream>>>(qraw, kraw,
                                                               vraw, qh, kh,
                                                               vh);

  attn_kernel<<<dim3(SEQL / 128, NBATCH * NHEADS), 256, 0, stream>>>(
      qh, kh, vh, amask, attn_out);

  quant_rows_kernel<8><<<MROWS, 256, 0, stream>>>(attn_out, aoq, aso);
  gemm_iu8_kernel<2><<<g1, 256, 0, stream>>>(aoq, aso, wqo, so, oa, x, x1,
                                             MROWS, HIDD, HIDD);

  rmsnorm_quant_kernel<<<MROWS, 256, 0, stream>>>(x1, n2, h2q, as2);
  dim3 g2(MLPDIM / 64, MROWS / 64);
  gemm_iu8_gated_kernel<<<g2, 256, 0, stream>>>(h2q, as2, wqg, sg, ga, wqu, su,
                                                ua, act, MROWS, MLPDIM, HIDD);
  quant_rows_kernel<32><<<MROWS, 256, 0, stream>>>(act, actq, as3);
  gemm_iu8_kernel<2><<<g1, 256, 0, stream>>>(actq, as3, wqd, sd, da, x1,
                                             (float*)d_out, MROWS, HIDD,
                                             MLPDIM);
}
